// AttentiveFP_54975581389399
// MI455X (gfx1250) — compile-verified, matrix operand order checked by reading, not run on hardware
//
#include <hip/hip_runtime.h>
#include <math.h>

#define NN 100000
#define EE 1600000
#define BB 2000
#define IN_C 39
#define HH 64
#define ED 10
#define K1 74

typedef __attribute__((ext_vector_type(16))) _Float16 v16h;
typedef __attribute__((ext_vector_type(8)))  _Float16 v8h;
typedef __attribute__((ext_vector_type(8)))  float    v8f;

__device__ __forceinline__ v8f wmma16(v16h a, v16h b, v8f c) {
  return __builtin_amdgcn_wmma_f32_16x16x32_f16(false, a, false, b, (short)0, c,
                                                false, false);
}

// A/B fragment (16-bit, 16x32) from a row-major f16 row pointer: per lane the
// fragment is two contiguous 8-half runs: [base .. base+7] and [base+16 .. base+23]
__device__ __forceinline__ v16h frag_f16row(const _Float16* __restrict__ rowp,
                                            int kc) {
  int lane = threadIdx.x & 31;
  int base = kc * 32 + ((lane & 16) ? 8 : 0);
  v8h lo = *(const v8h*)(rowp + base);
  v8h hi = *(const v8h*)(rowp + base + 16);
  v16h o;
#pragma unroll
  for (int e = 0; e < 8; ++e) {
    o[e] = lo[e];
    o[8 + e] = hi[e];
  }
  return o;
}

__device__ __forceinline__ v16h frag_f32row(const float* __restrict__ rowp,
                                            int kc) {
  int lane = threadIdx.x & 31;
  int base = kc * 32 + ((lane & 16) ? 8 : 0);
  v8f lo = *(const v8f*)(rowp + base);
  v8f hi = *(const v8f*)(rowp + base + 16);
  v16h o;
#pragma unroll
  for (int e = 0; e < 8; ++e) {
    o[e] = (_Float16)lo[e];
    o[8 + e] = (_Float16)hi[e];
  }
  return o;
}

// packed weight fragment (pre-swizzled by k_pack_w): one 32B contiguous load
__device__ __forceinline__ v16h load_packed(const _Float16* __restrict__ pk, int t,
                                            int kc, int nchunks) {
  int lane = threadIdx.x & 31;
  return *(const v16h*)&pk[(((size_t)t * nchunks + kc) * 32 + lane) * 16];
}

__device__ __forceinline__ float lrelu(float v) { return v > 0.f ? v : 0.01f * v; }
__device__ __forceinline__ float sigmf(float x) { return 1.f / (1.f + __expf(-x)); }

__device__ __forceinline__ unsigned f2ord(float f) {
  unsigned u = __float_as_uint(f);
  return (u & 0x80000000u) ? ~u : (u | 0x80000000u);
}
__device__ __forceinline__ float ord2f(unsigned s) {
  unsigned u = (s & 0x80000000u) ? (s & 0x7fffffffu) : ~s;
  return __uint_as_float(u);
}
#define ORD_NEG_INF 0x007FFFFFu

__global__ void k_fill_f32(float* p, long n, float v) {
  long i = (long)blockIdx.x * blockDim.x + threadIdx.x;
  if (i < n) p[i] = v;
}
__global__ void k_fill_u32(unsigned* p, long n, unsigned v) {
  long i = (long)blockIdx.x * blockDim.x + threadIdx.x;
  if (i < n) p[i] = v;
}
__global__ void k_relu_ip(float* p, long n) {
  long i = (long)blockIdx.x * blockDim.x + threadIdx.x;
  if (i < n) p[i] = fmaxf(p[i], 0.f);
}

// zero-padded f16 copy of x: (N x IN_C) f32 -> (N x 64) f16
__global__ void k_pad_x(const float* __restrict__ x, _Float16* __restrict__ xpad) {
  long i = (long)blockIdx.x * blockDim.x + threadIdx.x;
  if (i >= (long)NN * HH) return;
  int c = (int)(i & (HH - 1));
  long n = i >> 6;
  xpad[i] = (c < IN_C) ? (_Float16)x[n * IN_C + c] : (_Float16)0.f;
}

// zero-padded f16 copy of edge_attr: (E x 10) f32 -> (E x 32) f16
__global__ void k_pad_e(const float* __restrict__ ea, _Float16* __restrict__ epad) {
  long i = (long)blockIdx.x * blockDim.x + threadIdx.x;
  if (i >= (long)EE * 32) return;
  int c = (int)(i & 31);
  long e = i >> 5;
  epad[i] = (c < ED) ? (_Float16)ea[e * ED + c] : (_Float16)0.f;
}

// pack row-major weight W[cols_out x K] into WMMA B-fragment order (f16, zero pad K)
__global__ void k_pack_w(const float* __restrict__ W, _Float16* __restrict__ out,
                         int ncol_tiles, int nchunks, int K) {
  int idx = blockIdx.x * blockDim.x + threadIdx.x;
  int total = ncol_tiles * nchunks * 32;
  if (idx >= total) return;
  int lane = idx & 31;
  int kc = (idx >> 5) % nchunks;
  int t = (idx >> 5) / nchunks;
#pragma unroll
  for (int e = 0; e < 16; ++e) {
    int v = e >> 1;
    int kk = ((lane & 16) ? 8 : 0) + ((v & 4) ? 16 : 0) + ((v & 3) << 1) + (e & 1);
    int k = kc * 32 + kk;
    int c = t * 16 + (lane & 15);
    out[(size_t)idx * 16 + e] =
        (k < K) ? (_Float16)W[(size_t)c * K + k] : (_Float16)0.f;
  }
}

// x0 = leaky_relu(x @ W^T + b); also emits f16 mirror x0h for edge gathers
__global__ void __launch_bounds__(256)
k_lin1(const _Float16* __restrict__ xpad, const _Float16* __restrict__ pkW,
       const float* __restrict__ bias, float* __restrict__ y,
       _Float16* __restrict__ yh) {
  int tile = blockIdx.x * (blockDim.x >> 5) + (threadIdx.x >> 5);
  int row0 = tile * 16;
  if (row0 >= NN) return;
  int lane = threadIdx.x & 31;
  int myrow = lane & 15;
  const _Float16* rowp = xpad + (size_t)(row0 + myrow) * HH;
  v8f acc[4] = {};
#pragma unroll
  for (int kc = 0; kc < 2; ++kc) {
    v16h a = frag_f16row(rowp, kc);
#pragma unroll
    for (int t = 0; t < 4; ++t) acc[t] = wmma16(a, load_packed(pkW, t, kc, 2), acc[t]);
  }
  int col = lane & 15, rb = (lane & 16) ? 8 : 0;
#pragma unroll
  for (int t = 0; t < 4; ++t) {
    float bc = bias[t * 16 + col];
#pragma unroll
    for (int v = 0; v < 8; ++v) {
      float val = lrelu(acc[t][v] + bc);
      size_t o = (size_t)(row0 + rb + v) * HH + t * 16 + col;
      y[o] = val;
      yh[o] = (_Float16)val;
    }
  }
}

__global__ void __launch_bounds__(256)
k_gemm64(const float* __restrict__ xin, const _Float16* __restrict__ pkW,
         float* __restrict__ y, int nrows) {
  int tile = blockIdx.x * (blockDim.x >> 5) + (threadIdx.x >> 5);
  int row0 = tile * 16;
  if (row0 >= nrows) return;
  int lane = threadIdx.x & 31;
  int myrow = lane & 15;
  const float* rowp = xin + (size_t)(row0 + myrow) * HH;
  v8f acc[4] = {};
#pragma unroll
  for (int kc = 0; kc < 2; ++kc) {
    v16h a = frag_f32row(rowp, kc);
#pragma unroll
    for (int t = 0; t < 4; ++t) acc[t] = wmma16(a, load_packed(pkW, t, kc, 2), acc[t]);
  }
  int col = lane & 15, rb = (lane & 16) ? 8 : 0;
#pragma unroll
  for (int t = 0; t < 4; ++t)
#pragma unroll
    for (int v = 0; v < 8; ++v)
      y[(size_t)(row0 + rb + v) * HH + t * 16 + col] = acc[t][v];
}

// GATEConv pass A: xj = leaky(lin1([x_src, e_attr])), fused attention logits
__global__ void __launch_bounds__(256)
k_gate_xj(const _Float16* __restrict__ x0h, const _Float16* __restrict__ epad,
          const int* __restrict__ src, const int* __restrict__ dst,
          const _Float16* __restrict__ pkW1, const float* __restrict__ attl,
          const float* __restrict__ xr, _Float16* __restrict__ xj,
          float* __restrict__ elog, unsigned* __restrict__ mseg) {
  int tile = blockIdx.x * (blockDim.x >> 5) + (threadIdx.x >> 5);
  int e0 = tile * 16;
  if (e0 >= EE) return;
  int lane = threadIdx.x & 31;
  int myrow = lane & 15;
  int mysrc = src[e0 + myrow];
  const _Float16* xrow = x0h + (size_t)mysrc * HH;
  const _Float16* erow = epad + (size_t)(e0 + myrow) * 32;
  v8f acc[4] = {};
#pragma unroll
  for (int kc = 0; kc < 3; ++kc) {
    v16h a = (kc < 2) ? frag_f16row(xrow, kc) : frag_f16row(erow, 0);
#pragma unroll
    for (int t = 0; t < 4; ++t) acc[t] = wmma16(a, load_packed(pkW1, t, kc, 3), acc[t]);
  }
  int col = lane & 15, rb = (lane & 16) ? 8 : 0;
  float al[4];
#pragma unroll
  for (int t = 0; t < 4; ++t) al[t] = attl[t * 16 + col];
  float rowdot[8];
#pragma unroll
  for (int v = 0; v < 8; ++v) {
    float s = 0.f;
#pragma unroll
    for (int t = 0; t < 4; ++t) {
      float val = lrelu(acc[t][v]);
      acc[t][v] = val;
      s += val * al[t];
    }
    s += __shfl_xor(s, 1);
    s += __shfl_xor(s, 2);
    s += __shfl_xor(s, 4);
    s += __shfl_xor(s, 8);
    rowdot[v] = s;
  }
#pragma unroll
  for (int t = 0; t < 4; ++t)
#pragma unroll
    for (int v = 0; v < 8; ++v)
      xj[(size_t)(e0 + rb + v) * HH + t * 16 + col] = (_Float16)acc[t][v];
  if (col == 0) {
#pragma unroll
    for (int v = 0; v < 8; ++v) {
      int row = rb + v;
      int d = dst[e0 + row];
      float l = lrelu(rowdot[v] + xr[d]);
      elog[e0 + row] = l;
      atomicMax(&mseg[d], f2ord(l));
    }
  }
}

// GATEConv pass D: hn[dst] += alpha * (xj @ w2^T)
__global__ void __launch_bounds__(256)
k_gate_msg(const _Float16* __restrict__ xj, const _Float16* __restrict__ pkW2,
           const int* __restrict__ dst, const float* __restrict__ ev,
           const float* __restrict__ sseg, float* __restrict__ hacc) {
  int tile = blockIdx.x * (blockDim.x >> 5) + (threadIdx.x >> 5);
  int e0 = tile * 16;
  if (e0 >= EE) return;
  int lane = threadIdx.x & 31;
  int myrow = lane & 15;
  int down = dst[e0 + myrow];
  float aown = ev[e0 + myrow] / (sseg[down] + 1e-16f);
  const _Float16* rowp = xj + (size_t)(e0 + myrow) * HH;
  v8f acc[4] = {};
#pragma unroll
  for (int kc = 0; kc < 2; ++kc) {
    v16h a = frag_f16row(rowp, kc);
#pragma unroll
    for (int t = 0; t < 4; ++t) acc[t] = wmma16(a, load_packed(pkW2, t, kc, 2), acc[t]);
  }
  int col = lane & 15, rb = (lane & 16) ? 8 : 0;
#pragma unroll
  for (int v = 0; v < 8; ++v) {
    int row = rb + v;
    float alpha = __shfl(aown, row);
    int d = __shfl(down, row);
#pragma unroll
    for (int t = 0; t < 4; ++t)
      atomicAdd(&hacc[(size_t)d * HH + t * 16 + col], acc[t][v] * alpha);
  }
}

// fused GRU cell: out = relu(GRUCell(inp, hid)) row-tile per wave
__global__ void __launch_bounds__(256)
k_gru(const float* __restrict__ inp, const float* __restrict__ hid,
      const _Float16* __restrict__ pkWi, const _Float16* __restrict__ pkWh,
      const float* __restrict__ bi, const float* __restrict__ bh,
      float* __restrict__ outp, int nrows) {
  int tile = blockIdx.x * (blockDim.x >> 5) + (threadIdx.x >> 5);
  int row0 = tile * 16;
  if (row0 >= nrows) return;
  int lane = threadIdx.x & 31;
  int myrow = lane & 15;
  const float* irow = inp + (size_t)(row0 + myrow) * HH;
  const float* hrow = hid + (size_t)(row0 + myrow) * HH;
  v16h ai[2], ah[2];
#pragma unroll
  for (int kc = 0; kc < 2; ++kc) {
    ai[kc] = frag_f32row(irow, kc);
    ah[kc] = frag_f32row(hrow, kc);
  }
  int col = lane & 15, rb = (lane & 16) ? 8 : 0;
#pragma unroll
  for (int t = 0; t < 4; ++t) {
    int c = t * 16 + col;
    v8f accR = {}, accZ = {}, accNi = {}, accNh = {};
#pragma unroll
    for (int kc = 0; kc < 2; ++kc) {
      accR  = wmma16(ai[kc], load_packed(pkWi, 0 + t, kc, 2), accR);
      accR  = wmma16(ah[kc], load_packed(pkWh, 0 + t, kc, 2), accR);
      accZ  = wmma16(ai[kc], load_packed(pkWi, 4 + t, kc, 2), accZ);
      accZ  = wmma16(ah[kc], load_packed(pkWh, 4 + t, kc, 2), accZ);
      accNi = wmma16(ai[kc], load_packed(pkWi, 8 + t, kc, 2), accNi);
      accNh = wmma16(ah[kc], load_packed(pkWh, 8 + t, kc, 2), accNh);
    }
    float bir = bi[c], bhr = bh[c];
    float biz = bi[64 + c], bhz = bh[64 + c];
    float bin = bi[128 + c], bhn = bh[128 + c];
#pragma unroll
    for (int v = 0; v < 8; ++v) {
      int row = row0 + rb + v;
      float r = sigmf(accR[v] + bir + bhr);
      float z = sigmf(accZ[v] + biz + bhz);
      float n = tanhf(accNi[v] + bin + r * (accNh[v] + bhn));
      float hp = hid[(size_t)row * HH + c];
      float o2 = (1.f - z) * n + z * hp;
      outp[(size_t)row * HH + c] = fmaxf(o2, 0.f);
    }
  }
}

__global__ void k_node_dot(const float* __restrict__ xin,
                           const float* __restrict__ vec, float* __restrict__ out,
                           int n) {
  int i = blockIdx.x * blockDim.x + threadIdx.x;
  if (i >= n) return;
  float s = 0.f;
#pragma unroll 8
  for (int c = 0; c < HH; ++c) s += xin[(size_t)i * HH + c] * vec[c];
  out[i] = s;
}

__global__ void k_bias_elu(float* __restrict__ acc, const float* __restrict__ bias,
                           int nrows) {
  long i = (long)blockIdx.x * blockDim.x + threadIdx.x;
  if (i >= (long)nrows * HH) return;
  int c = (int)(i & (HH - 1));
  float v = acc[i] + bias[c];
  acc[i] = v > 0.f ? v : (__expf(v) - 1.f);
}

__global__ void k_edge_logit(const float* __restrict__ asn,
                             const float* __restrict__ adn,
                             const int* __restrict__ src, const int* __restrict__ dst,
                             float* __restrict__ elog, unsigned* __restrict__ m) {
  int e = blockIdx.x * blockDim.x + threadIdx.x;
  if (e >= EE) return;
  int d = dst[e];
  float l = lrelu(asn[src[e]] + adn[d]);
  elog[e] = l;
  atomicMax(&m[d], f2ord(l));
}

__global__ void k_edge_exp(float* __restrict__ elog, const int* __restrict__ dst,
                           const unsigned* __restrict__ m, float* __restrict__ s,
                           int nedges) {
  int e = blockIdx.x * blockDim.x + threadIdx.x;
  if (e >= nedges) return;
  int d = dst[e];
  float mv = ord2f(m[d]);
  if (!(mv > -1e37f)) mv = 0.f;
  float ex = __expf(elog[e] - mv);
  elog[e] = ex;
  atomicAdd(&s[d], ex);
}

// identity_src != 0 -> src index of edge e is e itself (bipartite node->molecule)
__global__ void k_edge_scatter(const float* __restrict__ elog,
                               const float* __restrict__ sseg,
                               const int* __restrict__ src, const int* __restrict__ dst,
                               const float* __restrict__ hsrc, float* __restrict__ hacc,
                               int nedges, int identity_src) {
  long idx = (long)blockIdx.x * blockDim.x + threadIdx.x;
  int e = (int)(idx >> 4);
  if (e >= nedges) return;
  int cg = (int)(idx & 15) * 4;
  int d = dst[e];
  int sidx = identity_src ? e : src[e];
  float w = elog[e] / (sseg[d] + 1e-16f);
  const float4 hv = *(const float4*)&hsrc[(size_t)sidx * HH + cg];
  atomicAdd(&hacc[(size_t)d * HH + cg + 0], w * hv.x);
  atomicAdd(&hacc[(size_t)d * HH + cg + 1], w * hv.y);
  atomicAdd(&hacc[(size_t)d * HH + cg + 2], w * hv.z);
  atomicAdd(&hacc[(size_t)d * HH + cg + 3], w * hv.w);
}

__global__ void k_seg_sum(const float* __restrict__ xin, const int* __restrict__ seg,
                          float* __restrict__ out) {
  long idx = (long)blockIdx.x * blockDim.x + threadIdx.x;
  int n = (int)(idx >> 4);
  if (n >= NN) return;
  int cg = (int)(idx & 15) * 4;
  int b = seg[n];
  const float4 v = *(const float4*)&xin[(size_t)n * HH + cg];
  atomicAdd(&out[(size_t)b * HH + cg + 0], v.x);
  atomicAdd(&out[(size_t)b * HH + cg + 1], v.y);
  atomicAdd(&out[(size_t)b * HH + cg + 2], v.z);
  atomicAdd(&out[(size_t)b * HH + cg + 3], v.w);
}

__global__ void k_mol_logit(const float* __restrict__ asn,
                            const float* __restrict__ adb,
                            const int* __restrict__ batch, float* __restrict__ elog,
                            unsigned* __restrict__ m) {
  int i = blockIdx.x * blockDim.x + threadIdx.x;
  if (i >= NN) return;
  int b = batch[i];
  float l = lrelu(asn[i] + adb[b]);
  elog[i] = l;
  atomicMax(&m[b], f2ord(l));
}

__global__ void k_head1(const float* __restrict__ out64, const float* __restrict__ w1,
                        const float* __restrict__ b1, float* __restrict__ hid) {
  int i = blockIdx.x * blockDim.x + threadIdx.x;
  if (i >= BB * 32) return;
  int b = i >> 5, c = i & 31;
  float s = b1[c];
#pragma unroll 8
  for (int k = 0; k < HH; ++k) s += out64[(size_t)b * HH + k] * w1[(size_t)c * HH + k];
  hid[i] = fmaxf(s, 0.f);
}

__global__ void k_head2(const float* __restrict__ hid, const float* __restrict__ w2,
                        const float* __restrict__ b2, float* __restrict__ y) {
  int b = blockIdx.x * blockDim.x + threadIdx.x;
  if (b >= BB) return;
  float s = b2[0];
#pragma unroll
  for (int k = 0; k < 32; ++k) s += hid[(size_t)b * 32 + k] * w2[k];
  y[b] = s;
}

extern "C" void kernel_launch(void* const* d_in, const int* in_sizes, int n_in,
                              void* d_out, int out_size, void* d_ws, size_t ws_size,
                              hipStream_t stream) {
  (void)in_sizes; (void)n_in; (void)out_size; (void)ws_size;
  const float* x_in     = (const float*)d_in[0];
  const int*   eidx     = (const int*)d_in[1];
  const float* eattr    = (const float*)d_in[2];
  const int*   batch    = (const int*)d_in[3];
  const float* lin1_w   = (const float*)d_in[4];
  const float* lin1_b   = (const float*)d_in[5];
  const float* ge_att_l = (const float*)d_in[6];
  const float* ge_att_r = (const float*)d_in[7];
  const float* ge_w1    = (const float*)d_in[8];
  const float* ge_w2    = (const float*)d_in[9];
  const float* ge_bias  = (const float*)d_in[10];
  const float* g0wi = (const float*)d_in[11];
  const float* g0wh = (const float*)d_in[12];
  const float* g0bi = (const float*)d_in[13];
  const float* g0bh = (const float*)d_in[14];
  const float* gat_w  = (const float*)d_in[15];
  const float* gat_as = (const float*)d_in[16];
  const float* gat_ad = (const float*)d_in[17];
  const float* gat_b  = (const float*)d_in[18];
  const float* g1wi = (const float*)d_in[19];
  const float* g1wh = (const float*)d_in[20];
  const float* g1bi = (const float*)d_in[21];
  const float* g1bh = (const float*)d_in[22];
  const float* mcs   = (const float*)d_in[23];
  const float* mcd   = (const float*)d_in[24];
  const float* mc_as = (const float*)d_in[25];
  const float* mc_ad = (const float*)d_in[26];
  const float* mc_b  = (const float*)d_in[27];
  const float* mgwi = (const float*)d_in[28];
  const float* mgwh = (const float*)d_in[29];
  const float* mgbi = (const float*)d_in[30];
  const float* mgbh = (const float*)d_in[31];
  const float* ph_w1 = (const float*)d_in[32];
  const float* ph_b1 = (const float*)d_in[33];
  const float* ph_w2 = (const float*)d_in[34];
  const float* ph_b2 = (const float*)d_in[35];
  const int* src  = eidx;
  const int* dstE = eidx + EE;
  float* y = (float*)d_out;

  float* Wb = (float*)d_ws;
  size_t o = 0;
  auto F = [&](size_t nfloats) {
    float* p = Wb + o;
    o += (nfloats + 7) & ~(size_t)7;
    return p;
  };
  float* x0  = F((size_t)NN * HH);
  float* x1  = F((size_t)NN * HH);
  float* hn  = F((size_t)NN * HH);
  float* hsb = F((size_t)NN * HH);
  float* ns0 = F(NN);
  float* ns1 = F(NN);
  unsigned* mN = (unsigned*)F(NN);
  float* sN  = F(NN);
  float* outA = F((size_t)BB * HH);
  float* outB = F((size_t)BB * HH);
  float* hd   = F((size_t)BB * HH);
  float* hb   = F((size_t)BB * HH);
  float* adb  = F(BB);
  unsigned* mB = (unsigned*)F(BB);
  float* sB  = F(BB);
  float* hid = F((size_t)BB * 32);
  float* ev  = F(EE);
  _Float16* x0h  = (_Float16*)F((size_t)NN * (HH / 2));   // f16 mirror of x0
  _Float16* xpad = (_Float16*)F((size_t)NN * (HH / 2));   // padded f16 input
  _Float16* epad = (_Float16*)F((size_t)EE * 16);          // E x 32 halves
  _Float16* xj   = (_Float16*)F((size_t)EE * (HH / 2));    // E x 64 halves
  _Float16* pk_lin1 = (_Float16*)F(4 * 2 * 256);
  _Float16* pk_gw1  = (_Float16*)F(4 * 3 * 256);
  _Float16* pk_gw2  = (_Float16*)F(4 * 2 * 256);
  _Float16* pk_g0wi = (_Float16*)F(12 * 2 * 256);
  _Float16* pk_g0wh = (_Float16*)F(12 * 2 * 256);
  _Float16* pk_gatw = (_Float16*)F(4 * 2 * 256);
  _Float16* pk_g1wi = (_Float16*)F(12 * 2 * 256);
  _Float16* pk_g1wh = (_Float16*)F(12 * 2 * 256);
  _Float16* pk_mcs  = (_Float16*)F(4 * 2 * 256);
  _Float16* pk_mcd  = (_Float16*)F(4 * 2 * 256);
  _Float16* pk_mgwi = (_Float16*)F(12 * 2 * 256);
  _Float16* pk_mgwh = (_Float16*)F(12 * 2 * 256);

  dim3 blk(256);
  auto g1 = [](long n) { return dim3((unsigned)((n + 255) / 256)); };
  auto gt = [](long tiles) { return dim3((unsigned)((tiles + 7) / 8)); };

  // weight packing + input padding (cheap, deterministic, every call)
  k_pack_w<<<g1(4 * 2 * 32), blk, 0, stream>>>(lin1_w, pk_lin1, 4, 2, IN_C);
  k_pack_w<<<g1(4 * 3 * 32), blk, 0, stream>>>(ge_w1, pk_gw1, 4, 3, K1);
  k_pack_w<<<g1(4 * 2 * 32), blk, 0, stream>>>(ge_w2, pk_gw2, 4, 2, HH);
  k_pack_w<<<g1(12 * 2 * 32), blk, 0, stream>>>(g0wi, pk_g0wi, 12, 2, HH);
  k_pack_w<<<g1(12 * 2 * 32), blk, 0, stream>>>(g0wh, pk_g0wh, 12, 2, HH);
  k_pack_w<<<g1(4 * 2 * 32), blk, 0, stream>>>(gat_w, pk_gatw, 4, 2, HH);
  k_pack_w<<<g1(12 * 2 * 32), blk, 0, stream>>>(g1wi, pk_g1wi, 12, 2, HH);
  k_pack_w<<<g1(12 * 2 * 32), blk, 0, stream>>>(g1wh, pk_g1wh, 12, 2, HH);
  k_pack_w<<<g1(4 * 2 * 32), blk, 0, stream>>>(mcs, pk_mcs, 4, 2, HH);
  k_pack_w<<<g1(4 * 2 * 32), blk, 0, stream>>>(mcd, pk_mcd, 4, 2, HH);
  k_pack_w<<<g1(12 * 2 * 32), blk, 0, stream>>>(mgwi, pk_mgwi, 12, 2, HH);
  k_pack_w<<<g1(12 * 2 * 32), blk, 0, stream>>>(mgwh, pk_mgwh, 12, 2, HH);
  k_pad_x<<<g1((long)NN * HH), blk, 0, stream>>>(x_in, xpad);
  k_pad_e<<<g1((long)EE * 32), blk, 0, stream>>>(eattr, epad);

  const long NT = NN / 16, ET = EE / 16, BT = BB / 16;

  k_lin1<<<gt(NT), blk, 0, stream>>>(xpad, pk_lin1, lin1_b, x0, x0h);

  // GATEConv
  k_node_dot<<<g1(NN), blk, 0, stream>>>(x0, ge_att_r, ns0, NN);
  k_fill_u32<<<g1(NN), blk, 0, stream>>>(mN, NN, ORD_NEG_INF);
  k_fill_f32<<<g1(NN), blk, 0, stream>>>(sN, NN, 0.f);
  k_gate_xj<<<gt(ET), blk, 0, stream>>>(x0h, epad, src, dstE, pk_gw1, ge_att_l,
                                        ns0, xj, ev, mN);
  k_edge_exp<<<g1(EE), blk, 0, stream>>>(ev, dstE, mN, sN, EE);
  k_fill_f32<<<g1((long)NN * HH), blk, 0, stream>>>(hn, (long)NN * HH, 0.f);
  k_gate_msg<<<gt(ET), blk, 0, stream>>>(xj, pk_gw2, dstE, ev, sN, hn);
  k_bias_elu<<<g1((long)NN * HH), blk, 0, stream>>>(hn, ge_bias, NN);
  k_gru<<<gt(NT), blk, 0, stream>>>(hn, x0, pk_g0wi, pk_g0wh, g0bi, g0bh, x1, NN);

  // GATConv
  k_gemm64<<<gt(NT), blk, 0, stream>>>(x1, pk_gatw, hsb, NN);
  k_node_dot<<<g1(NN), blk, 0, stream>>>(hsb, gat_as, ns0, NN);
  k_node_dot<<<g1(NN), blk, 0, stream>>>(hsb, gat_ad, ns1, NN);
  k_fill_u32<<<g1(NN), blk, 0, stream>>>(mN, NN, ORD_NEG_INF);
  k_fill_f32<<<g1(NN), blk, 0, stream>>>(sN, NN, 0.f);
  k_edge_logit<<<g1(EE), blk, 0, stream>>>(ns0, ns1, src, dstE, ev, mN);
  k_edge_exp<<<g1(EE), blk, 0, stream>>>(ev, dstE, mN, sN, EE);
  k_fill_f32<<<g1((long)NN * HH), blk, 0, stream>>>(hn, (long)NN * HH, 0.f);
  k_edge_scatter<<<g1((long)EE * 16), blk, 0, stream>>>(ev, sN, src, dstE, hsb, hn,
                                                        EE, 0);
  k_bias_elu<<<g1((long)NN * HH), blk, 0, stream>>>(hn, gat_b, NN);
  k_gru<<<gt(NT), blk, 0, stream>>>(hn, x1, pk_g1wi, pk_g1wh, g1bi, g1bh, x0, NN);

  // readout
  k_fill_f32<<<g1((long)BB * HH), blk, 0, stream>>>(outA, (long)BB * HH, 0.f);
  k_seg_sum<<<g1((long)NN * 16), blk, 0, stream>>>(x0, batch, outA);
  k_relu_ip<<<g1((long)BB * HH), blk, 0, stream>>>(outA, (long)BB * HH);

  // molecule attention timesteps
  k_gemm64<<<gt(NT), blk, 0, stream>>>(x0, pk_mcs, hsb, NN);
  k_node_dot<<<g1(NN), blk, 0, stream>>>(hsb, mc_as, ns0, NN);
  float* cur = outA;
  float* nxt = outB;
  for (int t = 0; t < 2; ++t) {
    k_gemm64<<<gt(BT), blk, 0, stream>>>(cur, pk_mcd, hd, BB);
    k_node_dot<<<g1(BB), blk, 0, stream>>>(hd, mc_ad, adb, BB);
    k_fill_u32<<<g1(BB), blk, 0, stream>>>(mB, BB, ORD_NEG_INF);
    k_fill_f32<<<g1(BB), blk, 0, stream>>>(sB, BB, 0.f);
    k_mol_logit<<<g1(NN), blk, 0, stream>>>(ns0, adb, batch, ev, mB);
    k_edge_exp<<<g1(NN), blk, 0, stream>>>(ev, batch, mB, sB, NN);
    k_fill_f32<<<g1((long)BB * HH), blk, 0, stream>>>(hb, (long)BB * HH, 0.f);
    k_edge_scatter<<<g1((long)NN * 16), blk, 0, stream>>>(ev, sB, src, batch, hsb,
                                                          hb, NN, 1);
    k_bias_elu<<<g1((long)BB * HH), blk, 0, stream>>>(hb, mc_b, BB);
    k_gru<<<gt(BT), blk, 0, stream>>>(hb, cur, pk_mgwi, pk_mgwh, mgbi, mgbh, nxt, BB);
    float* tmp = cur; cur = nxt; nxt = tmp;
  }

  k_head1<<<g1((long)BB * 32), blk, 0, stream>>>(cur, ph_w1, ph_b1, hid);
  k_head2<<<g1(BB), blk, 0, stream>>>(hid, ph_w2, ph_b2, y);
}
